// MultiHeadSelfAttention_300647711040
// MI455X (gfx1250) — matrix operand order repaired and checked by reading, not compiled, not run on hardware
//
#include <hip/hip_runtime.h>

// ---------------------------------------------------------------------------
// Multi-head self-attention forward for MI455X (gfx1250, wave32, WMMA).
//   convert x -> f16 ; transpose-convert Wqkv, Wout -> [N][K] f16
//   GEMM1: x @ Wqkv -> Q,K [B,H,L,hd] f16, V [B,H,hd,L] f16 (fused transposes)
//   flash-attn: WMMA QK^T + online softmax + WMMA PV -> attn [B,L,D] f16
//   GEMM2: attn @ Wout + b -> d_out fp32
// All matmuls: v_wmma_f32_16x16x32_f16, fp32 accumulation.
// Staging: GLOBAL_LOAD_ASYNC_TO_LDS_B128 into double-buffered LDS tiles,
// one barrier per K-step (fetch of tile t+1 overlaps WMMA of tile t).
// ---------------------------------------------------------------------------

typedef _Float16 h16;
typedef __attribute__((ext_vector_type(16))) _Float16 v16h;
typedef __attribute__((ext_vector_type(8)))  float    v8f;
typedef int v4i __attribute__((vector_size(16)));

#define DIMSZ   1024
#define NHEADS  16
#define HDIM    64
#define SEQ     2048
#define BATCH   2

union AF {
  v16h  v;
  h16   h[16];
  uint4 q[2];
};

// ---- async global->LDS copy (CDNA5), with sync fallback --------------------
#if __has_builtin(__builtin_amdgcn_global_load_async_to_lds_b128) && \
    __has_builtin(__builtin_amdgcn_s_wait_asynccnt)
#define HAS_ASYNC_LDS 1
typedef __attribute__((address_space(1))) v4i* gv4i_p;
typedef __attribute__((address_space(3))) v4i* lv4i_p;
__device__ __forceinline__ void cp16(h16* lds, const h16* g) {
  __builtin_amdgcn_global_load_async_to_lds_b128((gv4i_p)(g), (lv4i_p)(lds), 0, 0);
}
__device__ __forceinline__ void cp_wait() { __builtin_amdgcn_s_wait_asynccnt(0); }
#else
#define HAS_ASYNC_LDS 0
__device__ __forceinline__ void cp16(h16* lds, const h16* g) {
  *(uint4*)lds = *(const uint4*)g;
}
__device__ __forceinline__ void cp_wait() {}
#endif

// A-matrix 16x32 f16 (lane L: M = L%16; elem e: K = (e<8?e:e+8) + 8*(L/16)).
// Two aligned b128 LDS reads. Requires rowStride % 8 == 0, 16B-aligned base.
__device__ __forceinline__ v16h load_a_frag(const h16* base, int rowStride, int lane) {
  AF f;
  const h16* p = base + (lane & 15) * rowStride + (lane >> 4) * 8;
  f.q[0] = *(const uint4*)(p);        // K = kh + 0..7
  f.q[1] = *(const uint4*)(p + 16);   // K = kh + 16..23
  return f.v;
}

// B-matrix 32x16 f16 (lane L: N = L%16; elem e: K = (e<8?e:e+8) + 8*(L/16)) from a
// K-contiguous [N][K] tile (rowStride % 8 == 0).
__device__ __forceinline__ v16h load_b_fragT(const h16* base, int rowStride, int lane) {
  AF f;
  const h16* p = base + (lane & 15) * rowStride + (lane >> 4) * 8;
  f.q[0] = *(const uint4*)(p);
  f.q[1] = *(const uint4*)(p + 16);
  return f.v;
}

// ---------------------------------------------------------------------------
__global__ void f32_to_f16_v4(const float* __restrict__ src, h16* __restrict__ dst, int n) {
  int i = (blockIdx.x * blockDim.x + threadIdx.x) * 4;
  if (i >= n) return;
  float4 f = *(const float4*)(src + i);
  h16 tmp[4] = {(h16)f.x, (h16)f.y, (h16)f.z, (h16)f.w};
  *(uint2*)(dst + i) = *(uint2*)tmp;
}

// src[K][N] f32 -> dst[N][K] f16  (LDS-tiled 32x32 transpose, 256 threads)
__global__ __launch_bounds__(256) void transpose_f32_to_f16(
    const float* __restrict__ src, h16* __restrict__ dst, int K, int N) {
  __shared__ float tile[32][33];
  const int n0 = blockIdx.x * 32, k0 = blockIdx.y * 32;
  const int tx = threadIdx.x & 31, ty = threadIdx.x >> 5;   // 32 x 8
#pragma unroll
  for (int i = 0; i < 4; ++i)
    tile[ty + i * 8][tx] = src[(size_t)(k0 + ty + i * 8) * N + n0 + tx];
  __syncthreads();
#pragma unroll
  for (int i = 0; i < 4; ++i)
    dst[(size_t)(n0 + ty + i * 8) * K + k0 + tx] = (h16)tile[tx][ty + i * 8];
}

// ---------------------------------------------------------------------------
// f16 WMMA GEMM:  C[M,N] = A[M,K] @ BT[N,K]^T (+bias)
// Block tile 128x128, 8 waves (4 x 2), wave tile 32x64 = 2x4 WMMA tiles, K step 32.
// Double-buffered async staging; one barrier per K-step.
// mode 0: scatter -> Q,K [B,H,L,hd], V [B,H,hd,L] f16;  mode 1: fp32+bias -> Of
__global__ __launch_bounds__(256) void gemm_h16(
    const h16* __restrict__ A, const h16* __restrict__ BT,
    int M, int N, int K,
    const float* __restrict__ bias, int mode,
    h16* __restrict__ Qd, h16* __restrict__ Kd, h16* __restrict__ Vd,
    float* __restrict__ Of)
{
  __shared__ __align__(16) h16 Ash[2][128][40];   // [buf][m][k]
  __shared__ __align__(16) h16 Bsh[2][128][40];   // [buf][n][k]

  const int tid  = threadIdx.x;
  const int lane = tid & 31;
  const int wave = tid >> 5;
  const int mw   = wave & 3;    // 4 wave rows x 32
  const int nw   = wave >> 2;   // 2 wave cols x 64
  const int rowBase = blockIdx.y * 128;
  const int colBase = blockIdx.x * 128;

  v8f acc[2][4] = {};

  const int srow = tid >> 1, scol = (tid & 1) * 16;   // staging: 16 halfs/thread each
  const h16* agBase = A  + (size_t)(rowBase + srow) * K + scol;
  const h16* bgBase = BT + (size_t)(colBase + srow) * K + scol;

  // stage tile 0 into buffer 0
  cp16(&Ash[0][srow][scol],     agBase);
  cp16(&Ash[0][srow][scol + 8], agBase + 8);
  cp16(&Bsh[0][srow][scol],     bgBase);
  cp16(&Bsh[0][srow][scol + 8], bgBase + 8);

  int buf = 0;
  for (int kt = 0; kt < K; kt += 32) {
    cp_wait();             // tile kt resident (this thread's copies)
    __syncthreads();       // everyone staged kt; everyone done reading buf^1
    if (kt + 32 < K) {     // overlap: fetch tile kt+32 while computing kt
      const h16* ag = agBase + kt + 32;
      const h16* bg = bgBase + kt + 32;
      cp16(&Ash[buf ^ 1][srow][scol],     ag);
      cp16(&Ash[buf ^ 1][srow][scol + 8], ag + 8);
      cp16(&Bsh[buf ^ 1][srow][scol],     bg);
      cp16(&Bsh[buf ^ 1][srow][scol + 8], bg + 8);
      __builtin_prefetch(ag + 32, 0, 1);   // two tiles ahead
      __builtin_prefetch(bg + 32, 0, 1);
    }

    v16h af0 = load_a_frag(&Ash[buf][mw * 32][0],      40, lane);
    v16h af1 = load_a_frag(&Ash[buf][mw * 32 + 16][0], 40, lane);
#pragma unroll
    for (int j = 0; j < 4; ++j) {
      v16h bf = load_b_fragT(&Bsh[buf][nw * 64 + j * 16][0], 40, lane);
      acc[0][j] = __builtin_amdgcn_wmma_f32_16x16x32_f16(false, af0, false, bf, (short)0, acc[0][j], false, false);
      acc[1][j] = __builtin_amdgcn_wmma_f32_16x16x32_f16(false, af1, false, bf, (short)0, acc[1][j], false, false);
    }
    buf ^= 1;
  }

  // Epilogue. C layout: lane L -> N = L%16; VGPR r -> M = r + 8*(L/16).
  const int half  = lane >> 4;
  const int nn    = lane & 15;
  const int colHi = colBase + nw * 64;     // 64-aligned: uniform s / head per wave-col
  float bj[4];
#pragma unroll
  for (int j = 0; j < 4; ++j) bj[j] = bias[colHi + j * 16 + nn];

  if (mode == 0) {
    const int s  = colHi >> 10;                    // 0:Q 1:K 2:V
    const int hh = (colHi & (DIMSZ - 1)) >> 6;
    const int b  = rowBase >> 11;                  // block-uniform
    const int l0 = (rowBase & (SEQ - 1)) + mw * 32;
    if (s < 2) {                                   // Q,K: [B,H,L,hd]
      h16* dstb = ((s == 0) ? Qd : Kd) + ((size_t)b * NHEADS + hh) * SEQ * HDIM;
#pragma unroll
      for (int i = 0; i < 2; ++i)
#pragma unroll
        for (int r = 0; r < 8; ++r) {
          const int l = l0 + i * 16 + r + 8 * half;
#pragma unroll
          for (int j = 0; j < 4; ++j)
            dstb[(size_t)l * HDIM + j * 16 + nn] = (h16)(acc[i][j][r] + bj[j]);
        }
    } else {                                       // V: [B,H,hd,L]
      h16* dstb = Vd + ((size_t)b * NHEADS + hh) * HDIM * SEQ;
#pragma unroll
      for (int i = 0; i < 2; ++i)
#pragma unroll
        for (int r = 0; r < 8; ++r) {
          const int l = l0 + i * 16 + r + 8 * half;
#pragma unroll
          for (int j = 0; j < 4; ++j)
            dstb[(size_t)(j * 16 + nn) * SEQ + l] = (h16)(acc[i][j][r] + bj[j]);
        }
    }
  } else {
    float* ob = Of + (size_t)(rowBase + mw * 32) * N + colHi + nn;
#pragma unroll
    for (int i = 0; i < 2; ++i)
#pragma unroll
      for (int r = 0; r < 8; ++r)
#pragma unroll
        for (int j = 0; j < 4; ++j)
          ob[(size_t)(i * 16 + r + 8 * half) * N + j * 16] = acc[i][j][r] + bj[j];
  }
}

// ---------------------------------------------------------------------------
// Flash attention: block = 128 q rows of one (b,h); wave = 16 q rows.
// K is [B,H,L,hd]; V pre-transposed [B,H,hd,L]. Double-buffered K/V tiles.
__global__ __launch_bounds__(256) void attn_kernel(
    const h16* __restrict__ Qd, const h16* __restrict__ Kd, const h16* __restrict__ Vt,
    h16* __restrict__ Oh)   // [B, L, DIMSZ] f16
{
  __shared__ __align__(16) h16 Qsh[128][72];       // [q][hd]
  __shared__ __align__(16) h16 Ksh[2][32][72];     // [buf][key][hd]
  __shared__ __align__(16) h16 Vsh[2][64][40];     // [buf][hd][key]
  __shared__ __align__(16) h16 Psh[8][16][40];     // per-wave P roundtrip

  const int tid  = threadIdx.x;
  const int lane = tid & 31;
  const int wave = tid >> 5;
  const int bh   = blockIdx.y;                  // b*NHEADS + h
  const int b    = bh >> 4;
  const int h    = bh & (NHEADS - 1);
  const int qbase = blockIdx.x * 128;

  const h16* Qp = Qd + (size_t)bh * SEQ * HDIM;
  const h16* Kp = Kd + (size_t)bh * SEQ * HDIM;
  const h16* Vp = Vt + (size_t)bh * HDIM * SEQ;

  // Stage 128x64 Q tile once, pull persistent A-fragments.
  {
    const int r = tid >> 1, c = (tid & 1) * 32;
    const h16* p = Qp + (size_t)(qbase + r) * HDIM + c;
    cp16(&Qsh[r][c],      p);
    cp16(&Qsh[r][c + 8],  p + 8);
    cp16(&Qsh[r][c + 16], p + 16);
    cp16(&Qsh[r][c + 24], p + 24);
  }
  cp_wait();
  __syncthreads();
  v16h qa0 = load_a_frag(&Qsh[wave * 16][0],  72, lane);   // hd 0..31
  v16h qa1 = load_a_frag(&Qsh[wave * 16][32], 72, lane);   // hd 32..63

  v8f o[4] = {};
  float mrow[8], lrow[8];
#pragma unroll
  for (int r = 0; r < 8; ++r) { mrow[r] = -1e30f; lrow[r] = 0.0f; }

  const float scale = 0.125f;   // 1/sqrt(HDIM)
  const int krow = tid >> 3, kcol = (tid & 7) * 8;   // K tile: [32][64]
  const int vrow = tid >> 2, vcol = (tid & 3) * 8;   // V tile: [64][32]

  // stage K/V tile 0 into buffer 0
  cp16(&Ksh[0][krow][kcol], Kp + (size_t)krow * HDIM + kcol);
  cp16(&Vsh[0][vrow][vcol], Vp + (size_t)vrow * SEQ + vcol);

  int buf = 0;
  for (int kt = 0; kt < SEQ; kt += 32) {
    cp_wait();
    __syncthreads();
    if (kt + 32 < SEQ) {   // overlap next K/V tile fetch with this tile's math
      cp16(&Ksh[buf ^ 1][krow][kcol], Kp + (size_t)(kt + 32 + krow) * HDIM + kcol);
      cp16(&Vsh[buf ^ 1][vrow][vcol], Vp + (size_t)vrow * SEQ + kt + 32 + vcol);
    }

    // S[16 x 32] = scale * Q K^T
    v8f s0 = {}, s1 = {};
    {
      v16h kb = load_b_fragT(&Ksh[buf][0][0],   72, lane);
      s0 = __builtin_amdgcn_wmma_f32_16x16x32_f16(false, qa0, false, kb, (short)0, s0, false, false);
      kb      = load_b_fragT(&Ksh[buf][16][0],  72, lane);
      s1 = __builtin_amdgcn_wmma_f32_16x16x32_f16(false, qa0, false, kb, (short)0, s1, false, false);
      kb      = load_b_fragT(&Ksh[buf][0][32],  72, lane);
      s0 = __builtin_amdgcn_wmma_f32_16x16x32_f16(false, qa1, false, kb, (short)0, s0, false, false);
      kb      = load_b_fragT(&Ksh[buf][16][32], 72, lane);
      s1 = __builtin_amdgcn_wmma_f32_16x16x32_f16(false, qa1, false, kb, (short)0, s1, false, false);
    }

    // Online softmax. C layout: VGPR r -> row (r + 8*(lane/16)), col = lane%16.
    const int hl = lane >> 4;
    const int nn = lane & 15;
#pragma unroll
    for (int r = 0; r < 8; ++r) {
      float a = s0[r] * scale;
      float c = s1[r] * scale;
      float v = fmaxf(a, c);
      v = fmaxf(v, __shfl_xor(v, 1, 32));
      v = fmaxf(v, __shfl_xor(v, 2, 32));
      v = fmaxf(v, __shfl_xor(v, 4, 32));
      v = fmaxf(v, __shfl_xor(v, 8, 32));        // rowmax over this 32-key tile
      const float mn    = fmaxf(mrow[r], v);
      const float alpha = __expf(mrow[r] - mn);
      mrow[r] = mn;
      const float p0 = __expf(a - mn);
      const float p1 = __expf(c - mn);
      float rs = p0 + p1;
      rs += __shfl_xor(rs, 1, 32);
      rs += __shfl_xor(rs, 2, 32);
      rs += __shfl_xor(rs, 4, 32);
      rs += __shfl_xor(rs, 8, 32);
      lrow[r] = lrow[r] * alpha + rs;
      o[0][r] *= alpha; o[1][r] *= alpha; o[2][r] *= alpha; o[3][r] *= alpha;
      const int m = r + 8 * hl;
      Psh[wave][m][nn]      = (h16)p0;           // keys 0..15
      Psh[wave][m][16 + nn] = (h16)p1;           // keys 16..31
    }
    asm volatile("" ::: "memory");   // same-wave LDS is in-order; block reordering

    // O += P @ V
    v16h pa = load_a_frag(&Psh[wave][0][0], 40, lane);
#pragma unroll
    for (int t = 0; t < 4; ++t) {
      v16h vb = load_b_fragT(&Vsh[buf][t * 16][0], 40, lane);
      o[t] = __builtin_amdgcn_wmma_f32_16x16x32_f16(false, pa, false, vb, (short)0, o[t], false, false);
    }
    buf ^= 1;
  }

  // Normalize, write attn output in [b, l, h*64+hd] f16 layout.
  const int half = lane >> 4;
  const int nn   = lane & 15;
  float inv[8];
#pragma unroll
  for (int r = 0; r < 8; ++r) inv[r] = 1.0f / lrow[r];
  h16* ob = Oh + ((size_t)b * SEQ + qbase + wave * 16 + 8 * half) * DIMSZ + h * HDIM + nn;
#pragma unroll
  for (int t = 0; t < 4; ++t)
#pragma unroll
    for (int r = 0; r < 8; ++r)
      ob[(size_t)r * DIMSZ + t * 16] = (h16)(o[t][r] * inv[r]);
}

// ---------------------------------------------------------------------------
extern "C" void kernel_launch(void* const* d_in, const int* in_sizes, int n_in,
                              void* d_out, int out_size, void* d_ws, size_t ws_size,
                              hipStream_t stream) {
  const float* x    = (const float*)d_in[0];
  const float* Wqkv = (const float*)d_in[1];
  const float* bqkv = (const float*)d_in[2];
  const float* Wout = (const float*)d_in[3];
  const float* bout = (const float*)d_in[4];
  float* out = (float*)d_out;

  h16* ws = (h16*)d_ws;
  size_t off = 0;
  h16* xh     = ws + off; off += (size_t)BATCH * SEQ * DIMSZ;
  h16* wqkvT  = ws + off; off += (size_t)DIMSZ * 3 * DIMSZ;   // [3D][D]
  h16* woutT  = ws + off; off += (size_t)DIMSZ * DIMSZ;       // [D][D]
  h16* Qd     = ws + off; off += (size_t)BATCH * NHEADS * SEQ * HDIM;
  h16* Kd     = ws + off; off += (size_t)BATCH * NHEADS * SEQ * HDIM;
  h16* Vt     = ws + off; off += (size_t)BATCH * NHEADS * SEQ * HDIM;  // [B,H,hd,L]
  h16* attnh  = ws + off; off += (size_t)BATCH * SEQ * DIMSZ;
  (void)ws_size; (void)in_sizes; (void)n_in; (void)out_size;

  {
    int n = BATCH * SEQ * DIMSZ;
    f32_to_f16_v4<<<(n / 4 + 255) / 256, 256, 0, stream>>>(x, xh, n);
    transpose_f32_to_f16<<<dim3(3 * DIMSZ / 32, DIMSZ / 32), 256, 0, stream>>>(
        Wqkv, wqkvT, DIMSZ, 3 * DIMSZ);
    transpose_f32_to_f16<<<dim3(DIMSZ / 32, DIMSZ / 32), 256, 0, stream>>>(
        Wout, woutT, DIMSZ, DIMSZ);
  }

  // GEMM1: [B*L, D] @ Wqkv -> Q,K [B,H,L,hd], V [B,H,hd,L]
  {
    dim3 grid(3 * DIMSZ / 128, BATCH * SEQ / 128);
    gemm_h16<<<grid, 256, 0, stream>>>(xh, wqkvT, BATCH * SEQ, 3 * DIMSZ, DIMSZ,
                                       bqkv, 0, Qd, Kd, Vt, nullptr);
  }

  // Flash attention
  {
    dim3 grid(SEQ / 128, BATCH * NHEADS);
    attn_kernel<<<grid, 256, 0, stream>>>(Qd, Kd, Vt, attnh);
  }

  // GEMM2: [B*L, D] @ Wout + b_out -> fp32 d_out
  {
    dim3 grid(DIMSZ / 128, BATCH * SEQ / 128);
    gemm_h16<<<grid, 256, 0, stream>>>(attnh, woutT, BATCH * SEQ, DIMSZ, DIMSZ,
                                       bout, 1, nullptr, nullptr, nullptr, out);
  }
}
